// DeepRationalNeuralFingerprintHidden_42348377538856
// MI455X (gfx1250) — matrix operand order repaired and checked by reading, not compiled, not run on hardware
//
#include <hip/hip_runtime.h>
#include <hip/hip_bf16.h>
#include <stdint.h>

// ---------------------------------------------------------------------------
// DeepRationalNeuralFingerprintHidden for MI455X (gfx1250, wave32, WMMA).
//
// Shapes: B=2048, A=128, D=5, Fa=62, Fb=6, C=256.  Rows = B*A = 262144.
// FLOPs ~146 G; fused-intermediate bytes ~370 MB -> compute-bound on the
// matrix engine, so: bf16 WMMA (f32 accum), all 5 layers fused per 64-row
// tile with activations resident in LDS; weights (<0.6 MB bf16) stay in L2.
//
// Round-2 fixes: (a) force addrspace(1) on weight-fragment loads so they
// lower as global_load_b128 (pure LOADcnt) instead of flat_load_b128
// (LOADcnt+DScnt coupling that serialized WMMA issue); (b) fully unroll the
// layer loop so kdim/pointers are compile-time and k-loops unroll.
// ---------------------------------------------------------------------------

typedef __attribute__((ext_vector_type(16))) __bf16 v16bf;
typedef __attribute__((ext_vector_type(8)))  float  v8f;

// global (device memory) address space pointer for explicit global_load_*
typedef const v16bf __attribute__((address_space(1)))* gv16bfp;

#define NBATCH   2048
#define NATOMS   128
#define DEG      5
#define FA       62
#define FB       6
#define CH       256          // hidden width
#define K1       96           // 68 real K padded to 3 x 32
#define TM       64           // rows per workgroup
#define NROWS    (NBATCH * NATOMS)
#define HSTRIDE  512          // bytes per LDS activation row (256 bf16)
#define HBUF     (TM * HSTRIDE)   // 32 KiB per buffer, x2 = 64 KiB

// XOR-swizzled LDS offset (16-byte chunk granularity) for activation tiles:
// breaks bank collisions when 16 lanes read 16 different rows at one column,
// while keeping each 8-bf16 chunk contiguous (A-frag loads merge to b128).
__device__ __forceinline__ uint32_t h_off(int row, int col /*bf16 units*/) {
    uint32_t chunk = (uint32_t)(col >> 3) ^ (uint32_t)(row & 15);
    return (uint32_t)row * HSTRIDE + chunk * 16u + (uint32_t)(col & 7) * 2u;
}

__device__ __forceinline__ unsigned short f2bf_bits(float x) {
    return __builtin_bit_cast(unsigned short, (__bf16)x);
}

// ---------------------------------------------------------------------------
// Kernel 1: convert weights fp32 -> bf16, transposed to Wt[n][k] so a WMMA
// B-fragment (lane = output column n, 16 consecutive k) is one contiguous
// 32-byte global load. W1 is K-padded 68 -> 96 with zeros.
// ws layout (bf16 elems): Wt1[256*96] | Wt2..Wt5[256*256] each.
// ---------------------------------------------------------------------------
__global__ void nfp_prep_weights(const float* __restrict__ W1,
                                 const float* __restrict__ W2,
                                 const float* __restrict__ W3,
                                 const float* __restrict__ W4,
                                 const float* __restrict__ W5,
                                 unsigned short* __restrict__ wt) {
    const int N1  = CH * K1;                 // 24576
    const int TOT = N1 + 4 * CH * CH;        // 286720
    int idx = blockIdx.x * blockDim.x + threadIdx.x;
    if (idx >= TOT) return;
    if (idx < N1) {
        int n = idx / K1, k = idx % K1;
        float v = (k < FA + FB) ? W1[(size_t)k * CH + n] : 0.0f;
        wt[idx] = f2bf_bits(v);
    } else {
        int j = idx - N1;
        int l = j >> 16;                     // 0..3 -> W2..W5
        int r = j & 0xFFFF;
        int n = r >> 8, k = r & 255;
        const float* W = (l == 0) ? W2 : (l == 1) ? W3 : (l == 2) ? W4 : W5;
        wt[idx] = f2bf_bits(W[(size_t)k * CH + n]);
    }
}

// ---------------------------------------------------------------------------
// Kernel 2: fused aggregation + 5-layer rational MLP.
// Block: 256 threads (8 wave32), owns TM=64 rows. Grid: 262144/64 = 4096.
// Wave w: M-tile (w&3), N half ((w>>2)*128) -> 8 accumulator tiles of 16x16.
// ---------------------------------------------------------------------------
__global__ __launch_bounds__(256, 1)
void nfp_fused_mlp(const float* __restrict__ atoms,
                   const float* __restrict__ bonds,
                   const int*   __restrict__ edges,
                   const float* __restrict__ exist,
                   const unsigned short* __restrict__ wt,
                   const float* __restrict__ b1, const float* __restrict__ b2,
                   const float* __restrict__ b3, const float* __restrict__ b4,
                   const float* __restrict__ b5,
                   const float* __restrict__ alphas,
                   const float* __restrict__ betas,
                   float* __restrict__ out) {
    __shared__ __align__(16) unsigned char lds[2 * HBUF];   // 64 KiB

    const int tid     = threadIdx.x;
    const int rowBase = blockIdx.x * TM;

    // ---------------- stage 0: neighbor/bond aggregation -> LDS buf0 (bf16)
    {
        int lrow = tid >> 2;           // 0..63
        int part = tid & 3;            // 4 threads x 24 cols = 96 cols
        int rg   = rowBase + lrow;
        int molBase = (rg >> 7) << 7;  // b * 128
        int e[DEG];
#pragma unroll
        for (int d = 0; d < DEG; ++d) e[d] = edges[(size_t)rg * DEG + d];

        int c0 = part * 24;
        for (int f = c0; f < c0 + 24; ++f) {
            float v = 0.0f;
            if (f < FA) {
                v = atoms[(size_t)rg * FA + f];
#pragma unroll
                for (int d = 0; d < DEG; ++d)
                    if (e[d] >= 0) v += atoms[(size_t)(molBase + e[d]) * FA + f];
            } else if (f < FA + FB) {
#pragma unroll
                for (int d = 0; d < DEG; ++d)
                    v += bonds[((size_t)rg * DEG + d) * FB + (f - FA)];
            }
            *(unsigned short*)(lds + h_off(lrow, f)) = f2bf_bits(v);
        }
    }
    __syncthreads();

    // ---------------- 5 fused GEMM + rational-activation layers
    const int wave  = tid >> 5;
    const int lane  = tid & 31;
    const int ln    = lane & 15;       // column-within-tile / row-within-tile
    const int hi    = lane >> 4;       // half-wave select
    const int mt    = wave & 3;        // M tile 0..3 (rows mt*16..)
    const int nBase = (wave >> 2) * 128;

    const unsigned short* wts[5];
    wts[0] = wt;
    wts[1] = wts[0] + CH * K1;
    wts[2] = wts[1] + CH * CH;
    wts[3] = wts[2] + CH * CH;
    wts[4] = wts[3] + CH * CH;
    const float* biases[5] = {b1, b2, b3, b4, b5};

#pragma unroll
    for (int l = 0; l < 5; ++l) {
        const int kdim = (l == 0) ? K1 : CH;
        const unsigned char* src = lds + (l & 1) * HBUF;
        unsigned char*       dst = lds + ((l & 1) ^ 1) * HBUF;
        const unsigned short* W  = wts[l];
        const float* bias        = biases[l];

        // prefetch next layer's weight columns into L2/L0 while we compute
        if (l < 4)
            __builtin_prefetch(wts[l + 1] + (size_t)(nBase + ln) * CH, 0, 1);

        v8f acc[8] = {};

        for (int k0 = 0; k0 < kdim; k0 += 32) {
            // A fragment (16x32 bf16): lane ln = row; VGPR v:
            // k = k0 + 16*hi + 2*v   (pairs packed per dword)
            union { uint32_t u[8]; v16bf v; } A;
            const int arow = mt * 16 + ln;
#pragma unroll
            for (int v = 0; v < 8; ++v) {
                int k = k0 + (hi << 4) + (v << 1);
                A.u[v] = *(const uint32_t*)(src + h_off(arow, k));
            }
#pragma unroll
            for (int nt = 0; nt < 8; ++nt) {
                // B fragment (32x16 bf16): lane ln = column n; lanes 0-15
                // hold k0..k0+15, lanes 16-31 hold k0+16..k0+31 -> one
                // contiguous 32 B read from transposed bf16 weights.
                // Explicit addrspace(1) cast -> global_load_b128 (LOADcnt
                // only; keeps weight stream decoupled from LDS DScnt).
                int n = nBase + nt * 16 + ln;
                const v16bf Bv = *(gv16bfp)(const void*)(
                    W + (size_t)n * kdim + k0 + hi * 16);
                acc[nt] = __builtin_amdgcn_wmma_f32_16x16x32_bf16(
                    false, A.v, false, Bv, (short)0, acc[nt], false, false);
            }
        }

        // rational activation coefficients (uniform per layer)
        const float a0 = alphas[l * 4 + 0], a1 = alphas[l * 4 + 1];
        const float a2 = alphas[l * 4 + 2], a3 = alphas[l * 4 + 3];
        const float q0 = betas[l * 3 + 0],  q1 = betas[l * 3 + 1];
        const float q2 = betas[l * 3 + 2];

        // C/D layout: VGPR v -> M = v + 8*hi, N = ln
#pragma unroll
        for (int nt = 0; nt < 8; ++nt) {
            int col = nBase + nt * 16 + ln;
            float bc = bias[col];
#pragma unroll
            for (int v = 0; v < 8; ++v) {
                int row = mt * 16 + (hi << 3) + v;
                float x = acc[nt][v] + bc;
                float p = fmaf(fmaf(fmaf(a0, x, a1), x, a2), x, a3);
                float q = fmaf(fmaf(q0, x, q1), x, q2);
                float r = p / q;
                if (l == 4) {
                    int rg = rowBase + row;
                    out[(size_t)rg * CH + col] = r * exist[rg];
                } else {
                    *(unsigned short*)(dst + h_off(row, col)) = f2bf_bits(r);
                }
            }
        }
        __syncthreads();   // writes to dst visible before next layer reads it
    }
}

// ---------------------------------------------------------------------------
extern "C" void kernel_launch(void* const* d_in, const int* in_sizes, int n_in,
                              void* d_out, int out_size, void* d_ws, size_t ws_size,
                              hipStream_t stream) {
    const float* atoms  = (const float*)d_in[0];
    const float* bonds  = (const float*)d_in[1];
    const int*   edges  = (const int*)d_in[2];
    const float* exist  = (const float*)d_in[3];
    const float* W1     = (const float*)d_in[4];
    const float* b1     = (const float*)d_in[5];
    const float* W2     = (const float*)d_in[6];
    const float* b2     = (const float*)d_in[7];
    const float* W3     = (const float*)d_in[8];
    const float* b3     = (const float*)d_in[9];
    const float* W4     = (const float*)d_in[10];
    const float* b4     = (const float*)d_in[11];
    const float* W5     = (const float*)d_in[12];
    const float* b5     = (const float*)d_in[13];
    const float* alphas = (const float*)d_in[14];
    const float* betas  = (const float*)d_in[15];

    unsigned short* wt = (unsigned short*)d_ws;   // needs 573,440 bytes

    const int totW = CH * K1 + 4 * CH * CH;       // 286720 bf16 elems
    nfp_prep_weights<<<(totW + 255) / 256, 256, 0, stream>>>(W1, W2, W3, W4, W5, wt);

    nfp_fused_mlp<<<NROWS / TM, 256, 0, stream>>>(
        atoms, bonds, edges, exist, wt,
        b1, b2, b3, b4, b5, alphas, betas, (float*)d_out);
}
